// GRADUATE_3530463117925
// MI455X (gfx1250) — matrix operand order repaired and checked by reading, not compiled, not run on hardware
//
#include <hip/hip_runtime.h>
#include <hip/hip_bf16.h>

#define B_ 4
#define L_ 2048
#define D_ 256
#define H_ 8
#define DK_ 32
#define TOPK_ 64

typedef __bf16 v16bf __attribute__((ext_vector_type(16)));
typedef float  v8f   __attribute__((ext_vector_type(8)));

union BF16Frag { v16bf v; __bf16 h[16]; unsigned short u[16]; uint4 q[2]; };

// Native hardware convert (v_cvt_pk_bf16_f32 / v_cvt bf16 path), RNE like the ref.
__device__ __forceinline__ unsigned short f2b(float x) {
    __bf16 h = (__bf16)x;
    return __builtin_bit_cast(unsigned short, h);
}

__device__ __forceinline__ v8f bwmma(v16bf a, v16bf b, v8f c) {
    // D = A(16x32 bf16) * B(32x16 bf16) + C(16x16 f32)
    return __builtin_amdgcn_wmma_f32_16x16x32_bf16(false, a, false, b, (short)0, c, false, false);
}

// Build A-fragment (16x32, lane m = row) from 32 f32 per row, converting to bf16.
// p points at rowbase + kb + klaneA; pattern: halves 0..7 at +0, 8..15 at +16.
__device__ __forceinline__ v16bf load_a_from_f32(const float* p) {
    BF16Frag a;
    float4 x0 = *(const float4*)(p);
    float4 x1 = *(const float4*)(p + 4);
    float4 x2 = *(const float4*)(p + 16);
    float4 x3 = *(const float4*)(p + 20);
    a.h[0]  = (__bf16)x0.x; a.h[1]  = (__bf16)x0.y; a.h[2]  = (__bf16)x0.z; a.h[3]  = (__bf16)x0.w;
    a.h[4]  = (__bf16)x1.x; a.h[5]  = (__bf16)x1.y; a.h[6]  = (__bf16)x1.z; a.h[7]  = (__bf16)x1.w;
    a.h[8]  = (__bf16)x2.x; a.h[9]  = (__bf16)x2.y; a.h[10] = (__bf16)x2.z; a.h[11] = (__bf16)x2.w;
    a.h[12] = (__bf16)x3.x; a.h[13] = (__bf16)x3.y; a.h[14] = (__bf16)x3.z; a.h[15] = (__bf16)x3.w;
    return a.v;
}

// A-fragment from bf16 source (two contiguous 8-half runs at +0 and +16)
__device__ __forceinline__ v16bf load_a_from_b16(const unsigned short* p) {
    BF16Frag a;
    a.q[0] = *(const uint4*)(p);
    a.q[1] = *(const uint4*)(p + 16);
    return a.v;
}

// B-fragment: two contiguous 8-half runs at +0 and +16 (lane-half K offset already in p)
__device__ __forceinline__ v16bf load_b_b16(const unsigned short* p) {
    BF16Frag b;
    b.q[0] = *(const uint4*)(p);
    b.q[1] = *(const uint4*)(p + 16);
    return b.v;
}

// ---------------- Kernel 1: f32 -> bf16 weight convert ----------------
__global__ void k_cvt(const float* __restrict__ w, unsigned short* __restrict__ o, int n) {
    int i = blockIdx.x * blockDim.x + threadIdx.x;
    if (i < n) o[i] = f2b(w[i]);
}

// ---------------- Kernel 2: QKV projection ----------------
// One wave computes a 16-row x 32-col (one head) tile of X @ W^T.
// transposedV: store as [bh][dk][L] (for V), else [bh][L][dk].
__global__ __launch_bounds__(256) void k_proj(const float* __restrict__ x,
                                              const unsigned short* __restrict__ w16,
                                              unsigned short* __restrict__ out,
                                              int transposedV) {
    const int tid = threadIdx.x;
    const int wave = tid >> 5, lane = tid & 31;
    const int wid = blockIdx.x * 8 + wave;      // 0..4095
    const int h = wid & 7;
    const int qt = (wid >> 3) & 127;
    const int bb = wid >> 10;
    const int row0 = qt * 16;
    const int m = lane & 15;
    const int klaneA = (lane >> 4) * 8;
    const int nlane = lane & 15;
    const int kblaneB = (lane >> 4) * 8;
    const int Moff = (lane >> 4) * 8;

    v8f c0 = {0.f, 0.f, 0.f, 0.f, 0.f, 0.f, 0.f, 0.f};
    v8f c1 = c0;

    const float* xrow = x + (size_t)(bb * L_ + row0 + m) * D_;
#pragma unroll
    for (int kc = 0; kc < 8; ++kc) {
        const int kb = kc * 32;
        v16bf a = load_a_from_f32(xrow + kb + klaneA);
        const unsigned short* pb0 = w16 + (size_t)(h * 32 + nlane) * D_ + kb + kblaneB;
        const unsigned short* pb1 = w16 + (size_t)(h * 32 + 16 + nlane) * D_ + kb + kblaneB;
        c0 = bwmma(a, load_b_b16(pb0), c0);
        c1 = bwmma(a, load_b_b16(pb1), c1);
    }

    const int bh = bb * H_ + h;
#pragma unroll
    for (int r = 0; r < 8; ++r) {
        const int M = r + Moff;
        if (transposedV) {
            out[((size_t)bh * DK_ + nlane) * L_ + row0 + M]      = f2b(c0[r]);
            out[((size_t)bh * DK_ + 16 + nlane) * L_ + row0 + M] = f2b(c1[r]);
        } else {
            out[((size_t)bh * L_ + row0 + M) * DK_ + nlane]      = f2b(c0[r]);
            out[((size_t)bh * L_ + row0 + M) * DK_ + 16 + nlane] = f2b(c1[r]);
        }
    }
}

// ---------------- Kernel 3: attention with top-k threshold ----------------
// One block (8 waves) handles 16 query rows of one (b,h). Scores stay in LDS.
__global__ __launch_bounds__(256) void k_attn(const unsigned short* __restrict__ Qh,
                                              const unsigned short* __restrict__ Kh,
                                              const unsigned short* __restrict__ VhT,
                                              unsigned short* __restrict__ AttnO) {
    __shared__ float sc[16][2048];          // 128 KB score / weight block
    __shared__ float red[16][16];
    __shared__ float rowmax[16], rowlo[16], rowhi[16], rowsum[16];
    __shared__ float acc[16][DK_];

    const int tid = threadIdx.x, wave = tid >> 5, lane = tid & 31;
    const int qt = blockIdx.x & 127;
    const int h = (blockIdx.x >> 7) & 7;
    const int bb = blockIdx.x >> 10;
    const int bh = bb * H_ + h;
    const int row0 = qt * 16;
    const int m = lane & 15, klaneA = (lane >> 4) * 8;
    const int nlane = lane & 15, kblaneB = (lane >> 4) * 8;
    const int Moff = (lane >> 4) * 8;

    // Q fragment lives in registers for the whole score loop
    const v16bf aq = load_a_from_b16(Qh + ((size_t)bh * L_ + row0 + m) * DK_ + klaneA);

    // scores: Q(16x32) x Kh^T tiles -> LDS
    for (int t = wave; t < 128; t += 8) {
        const int key0 = t * 16;
        const unsigned short* pk = Kh + ((size_t)bh * L_ + key0 + nlane) * DK_ + kblaneB;
        if (t + 8 < 128) __builtin_prefetch(pk + 8 * 16 * DK_, 0, 1);
        v8f c = {0.f, 0.f, 0.f, 0.f, 0.f, 0.f, 0.f, 0.f};
        c = bwmma(aq, load_b_b16(pk), c);
#pragma unroll
        for (int r = 0; r < 8; ++r)
            sc[r + Moff][key0 + nlane] = c[r] * 0.17677669529663687f;  // 1/sqrt(dk)
    }
    float* accf = &acc[0][0];
    accf[tid] = 0.f; accf[tid + 256] = 0.f;
    __syncthreads();

    const int row = tid >> 4, j0 = tid & 15;

    // row max
    float lm = -1e30f;
    for (int j = j0; j < 2048; j += 16) lm = fmaxf(lm, sc[row][j]);
    red[row][j0] = lm;
    __syncthreads();
    if (tid < 16) {
        float mx = red[tid][0];
#pragma unroll
        for (int jj = 1; jj < 16; ++jj) mx = fmaxf(mx, red[tid][jj]);
        rowmax[tid] = mx; rowlo[tid] = mx - 30.f; rowhi[tid] = mx + 1.f;
    }
    __syncthreads();

    // binary search for the 64th-largest score (threshold); softmax is monotone
    for (int it = 0; it < 30; ++it) {
        const float mid = 0.5f * (rowlo[row] + rowhi[row]);
        float cnt = 0.f;
        for (int j = j0; j < 2048; j += 16) cnt += (sc[row][j] >= mid) ? 1.f : 0.f;
        red[row][j0] = cnt;
        __syncthreads();
        if (tid < 16) {
            float tot = 0.f;
#pragma unroll
            for (int jj = 0; jj < 16; ++jj) tot += red[tid][jj];
            const float m2 = 0.5f * (rowlo[tid] + rowhi[tid]);
            if (tot >= (float)TOPK_) rowlo[tid] = m2; else rowhi[tid] = m2;
        }
        __syncthreads();
    }

    // masked exp in place + row sums
    {
        const float th = rowlo[row], mx = rowmax[row];
        float ls = 0.f;
        for (int j = j0; j < 2048; j += 16) {
            const float s = sc[row][j];
            const float w = (s >= th) ? __expf(s - mx) : 0.f;
            sc[row][j] = w; ls += w;
        }
        red[row][j0] = ls;
        __syncthreads();
        if (tid < 16) {
            float tot = 0.f;
#pragma unroll
            for (int jj = 0; jj < 16; ++jj) tot += red[tid][jj];
            rowsum[tid] = tot;
        }
        __syncthreads();
    }

    // out = W @ V : each wave takes 8 K-chunks of 32 keys
    v8f c0 = {0.f, 0.f, 0.f, 0.f, 0.f, 0.f, 0.f, 0.f};
    v8f c1 = c0;
#pragma unroll
    for (int i = 0; i < 8; ++i) {
        const int kb = (wave * 8 + i) * 32;
        v16bf a = load_a_from_f32(&sc[m][kb + klaneA]);
        const unsigned short* pv0 = VhT + ((size_t)bh * DK_ + nlane) * L_ + kb + kblaneB;
        const unsigned short* pv1 = VhT + ((size_t)bh * DK_ + 16 + nlane) * L_ + kb + kblaneB;
        c0 = bwmma(a, load_b_b16(pv0), c0);
        c1 = bwmma(a, load_b_b16(pv1), c1);
    }
#pragma unroll
    for (int r = 0; r < 8; ++r) {
        atomicAdd(&acc[r + Moff][nlane], c0[r]);        // ds_add_f32 cross-wave reduce
        atomicAdd(&acc[r + Moff][16 + nlane], c1[r]);
    }
    __syncthreads();

    // renormalize (fold 1/rowsum) and emit bf16 head slice
    int idx = tid;
#pragma unroll
    for (int rep = 0; rep < 2; ++rep, idx += 256) {
        const int rr = idx >> 5, cc = idx & 31;
        const float v = acc[rr][cc] / rowsum[rr];
        AttnO[(size_t)(bb * L_ + row0 + rr) * D_ + h * DK_ + cc] = f2b(v);
    }
}

// ---------------- Kernel 4: fc projection + residual + LayerNorm ----------------
__global__ __launch_bounds__(256) void k_fc_ln(const unsigned short* __restrict__ AttnO,
                                               const unsigned short* __restrict__ Wfc,
                                               const float* __restrict__ qin,
                                               const float* __restrict__ gamma,
                                               const float* __restrict__ beta,
                                               float* __restrict__ out) {
    __shared__ float obuf[16][256];
    __shared__ float red[16][16];
    __shared__ float mu[16], rs[16];

    const int tid = threadIdx.x, wave = tid >> 5, lane = tid & 31;
    const int qt = blockIdx.x & 127, bb = blockIdx.x >> 7;
    const int row0 = qt * 16;
    const int m = lane & 15, klaneA = (lane >> 4) * 8;
    const int nlane = lane & 15, kblaneB = (lane >> 4) * 8;
    const int Moff = (lane >> 4) * 8;
    const int n0 = wave * 32;

    v8f c0 = {0.f, 0.f, 0.f, 0.f, 0.f, 0.f, 0.f, 0.f};
    v8f c1 = c0;
    const unsigned short* arow = AttnO + (size_t)(bb * L_ + row0 + m) * D_;
#pragma unroll
    for (int kc = 0; kc < 8; ++kc) {
        const int kb = kc * 32;
        v16bf a = load_a_from_b16(arow + kb + klaneA);
        const unsigned short* pb0 = Wfc + (size_t)(n0 + nlane) * D_ + kb + kblaneB;
        const unsigned short* pb1 = Wfc + (size_t)(n0 + 16 + nlane) * D_ + kb + kblaneB;
        c0 = bwmma(a, load_b_b16(pb0), c0);
        c1 = bwmma(a, load_b_b16(pb1), c1);
    }
#pragma unroll
    for (int r = 0; r < 8; ++r) {
        const int M = r + Moff;
        const float* qrow = qin + (size_t)(bb * L_ + row0 + M) * D_;
        obuf[M][n0 + nlane]      = c0[r] + qrow[n0 + nlane];
        obuf[M][n0 + 16 + nlane] = c1[r] + qrow[n0 + 16 + nlane];
    }
    __syncthreads();

    const int row = tid >> 4, j0 = tid & 15;
    float s = 0.f, ss = 0.f;
    for (int j = j0; j < 256; j += 16) { const float x = obuf[row][j]; s += x; ss += x * x; }
    red[row][j0] = s;
    __syncthreads();
    if (tid < 16) {
        float t = 0.f;
#pragma unroll
        for (int jj = 0; jj < 16; ++jj) t += red[tid][jj];
        mu[tid] = t * (1.f / 256.f);
    }
    __syncthreads();
    red[row][j0] = ss;
    __syncthreads();
    if (tid < 16) {
        float t = 0.f;
#pragma unroll
        for (int jj = 0; jj < 16; ++jj) t += red[tid][jj];
        const float var = t * (1.f / 256.f) - mu[tid] * mu[tid];
        rs[tid] = rsqrtf(var + 1e-6f);
    }
    __syncthreads();
    const float mym = mu[row], myr = rs[row];
    for (int j = j0; j < 256; j += 16) {
        const float x = obuf[row][j];
        out[(size_t)(bb * L_ + row0 + row) * D_ + j] = (x - mym) * myr * gamma[j] + beta[j];
    }
}

extern "C" void kernel_launch(void* const* d_in, const int* in_sizes, int n_in,
                              void* d_out, int out_size, void* d_ws, size_t ws_size,
                              hipStream_t stream) {
    const float* q     = (const float*)d_in[0];
    const float* k     = (const float*)d_in[1];
    const float* v     = (const float*)d_in[2];
    const float* w_q   = (const float*)d_in[3];
    const float* w_k   = (const float*)d_in[4];
    const float* w_v   = (const float*)d_in[5];
    const float* w_fc  = (const float*)d_in[6];
    const float* gamma = (const float*)d_in[7];
    const float* beta  = (const float*)d_in[8];
    float* out = (float*)d_out;

    const size_t nQ = (size_t)B_ * H_ * L_ * DK_;        // 2,097,152
    unsigned short* Qh    = (unsigned short*)d_ws;
    unsigned short* Kh    = Qh + nQ;
    unsigned short* VhT   = Kh + nQ;
    unsigned short* Wq16  = VhT + nQ;
    unsigned short* Wk16  = Wq16 + D_ * D_;
    unsigned short* Wv16  = Wk16 + D_ * D_;
    unsigned short* Wfc16 = Wv16 + D_ * D_;
    unsigned short* AttnO = Wfc16 + D_ * D_;             // B*L*D bf16

    k_cvt<<<256, 256, 0, stream>>>(w_q, Wq16, D_ * D_);
    k_cvt<<<256, 256, 0, stream>>>(w_k, Wk16, D_ * D_);
    k_cvt<<<256, 256, 0, stream>>>(w_v, Wv16, D_ * D_);
    k_cvt<<<256, 256, 0, stream>>>(w_fc, Wfc16, D_ * D_);

    k_proj<<<512, 256, 0, stream>>>(q, Wq16, Qh, 0);
    k_proj<<<512, 256, 0, stream>>>(k, Wk16, Kh, 0);
    k_proj<<<512, 256, 0, stream>>>(v, Wv16, VhT, 1);

    k_attn<<<B_ * H_ * (L_ / 16), 256, 0, stream>>>(Qh, Kh, VhT, AttnO);

    k_fc_ln<<<B_ * (L_ / 16), 256, 0, stream>>>(AttnO, Wfc16, q, gamma, beta, out);
}
